// Delta_43052752175269
// MI455X (gfx1250) — matrix operand order repaired and checked by reading, not compiled, not run on hardware
//
#include <hip/hip_runtime.h>

// Sizes from the reference
#define M_TOT   512      // B*S = 2*256
#define N_OUT   8192
#define K_IN    8192
#define RANK    1024

#define THREADS 256      // 8 wave32 waves
#define BM 128
#define BN 128
#define BK 64
#define LDSP (BK + 8)                  // 72 halves = 144B padded LDS row stride
#define NVA (BM * BK / 4 / THREADS)    // 8 float4/thread for A tile
#define NVB (BN * BK / 4 / THREADS)    // 8 float4/thread for B tile
#define SMEM_BYTES ((2 * BM * LDSP + 2 * BN * LDSP) * (int)sizeof(_Float16))  // 73728 B dynamic LDS

typedef _Float16 v16h __attribute__((ext_vector_type(16)));
typedef _Float16 v8h  __attribute__((ext_vector_type(8)));
typedef _Float16 h4   __attribute__((ext_vector_type(4)));
typedef float    v8f  __attribute__((ext_vector_type(8)));
typedef float    f4   __attribute__((ext_vector_type(4)));

// ---- WMMA fragment loads from LDS (layouts per CDNA5 ISA 7.12.2) ----
// A 16x32 f16: lanes 0-15 -> M=lane, K halves {0..7,16..23}; lanes 16-31 -> K {8..15,24..31}
__device__ __forceinline__ v16h frag_a(const _Float16* s, int lane, int row, int kb) {
    const _Float16* p = s + row * LDSP + kb + ((lane >> 4) << 3);
    v8h lo = *(const v8h*)p;
    v8h hi = *(const v8h*)(p + 16);
    return __builtin_shufflevector(lo, hi, 0,1,2,3,4,5,6,7,8,9,10,11,12,13,14,15);
}
// B 32x16 f16: lanes 0-15 -> N=lane, K halves {0..7,16..23}; lanes 16-31 -> K {8..15,24..31}
__device__ __forceinline__ v16h frag_b(const _Float16* s, int lane, int col, int kb) {
    const _Float16* p = s + col * LDSP + kb + ((lane >> 4) << 3);
    v8h lo = *(const v8h*)p;
    v8h hi = *(const v8h*)(p + 16);
    return __builtin_shufflevector(lo, hi, 0,1,2,3,4,5,6,7,8,9,10,11,12,13,14,15);
}

// One K=32 step for a 64x32 wave tile: 4 A-frags x 2 B-frags -> 8 WMMAs
__device__ __forceinline__ void mma_step(const _Float16* sA, const _Float16* sB,
                                         int lane, int mrow, int ncol, int ks,
                                         v8f (&acc)[4][2]) {
    const int kb = ks * 32;
    v16h b0 = frag_b(sB, lane, ncol +      (lane & 15), kb);
    v16h b1 = frag_b(sB, lane, ncol + 16 + (lane & 15), kb);
    #pragma unroll
    for (int fm = 0; fm < 4; ++fm) {
        v16h a = frag_a(sA, lane, mrow + fm * 16 + (lane & 15), kb);
        acc[fm][0] = __builtin_amdgcn_wmma_f32_16x16x32_f16(false, a, false, b0, (short)0, acc[fm][0], false, false);
        acc[fm][1] = __builtin_amdgcn_wmma_f32_16x16x32_f16(false, a, false, b1, (short)0, acc[fm][1], false, false);
    }
}

// ---- fp32 -> f16 LDS staging ----
__device__ __forceinline__ void stash_a(_Float16* sA, const f4 (&r)[NVA], int tid) {
    #pragma unroll
    for (int i = 0; i < NVA; ++i) {
        int idx = tid + i * THREADS, row = idx >> 4, c = idx & 15;
        *(h4*)(sA + row * LDSP + c * 4) = __builtin_convertvector(r[i], h4);
    }
}
__device__ __forceinline__ void stash_b(_Float16* sB, const f4 (&r)[NVB], int tid) {
    #pragma unroll
    for (int i = 0; i < NVB; ++i) {
        int idx = tid + i * THREADS, row = idx >> 4, c = idx & 15;
        *(h4*)(sB + row * LDSP + c * 4) = __builtin_convertvector(r[i], h4);
    }
}

// One pipelined GEMM pass: acc += f16(A[m0.., :ksize]) * f16(W[n0.., :ksize])^T
// A: [M, lda] row-major activations; W: [N, ldw] row-major weights.
__device__ __forceinline__ void gemm_pass(const float* __restrict__ A, int lda,
                                          const float* __restrict__ W, int ldw,
                                          int ksize, int m0, int n0,
                                          _Float16* sA, _Float16* sB,
                                          int tid, int lane, int mrow, int ncol,
                                          v8f (&acc)[4][2]) {
    f4 ra[NVA], rb[NVB];
    #pragma unroll
    for (int i = 0; i < NVA; ++i) {
        int idx = tid + i * THREADS, row = idx >> 4, c = idx & 15;
        ra[i] = *(const f4*)(A + (size_t)(m0 + row) * lda + c * 4);
    }
    #pragma unroll
    for (int i = 0; i < NVB; ++i) {
        int idx = tid + i * THREADS, row = idx >> 4, c = idx & 15;
        rb[i] = *(const f4*)(W + (size_t)(n0 + row) * ldw + c * 4);
    }
    stash_a(sA, ra, tid);
    stash_b(sB, rb, tid);
    __syncthreads();

    const int nt = ksize / BK;
    int cur = 0;
    for (int t = 0; t < nt; ++t) {
        if (t + 1 < nt) {  // issue next-stage global loads; WMMAs below overlap them
            const int k0 = (t + 1) * BK;
            #pragma unroll
            for (int i = 0; i < NVA; ++i) {
                int idx = tid + i * THREADS, row = idx >> 4, c = idx & 15;
                ra[i] = *(const f4*)(A + (size_t)(m0 + row) * lda + k0 + c * 4);
            }
            #pragma unroll
            for (int i = 0; i < NVB; ++i) {
                int idx = tid + i * THREADS, row = idx >> 4, c = idx & 15;
                rb[i] = *(const f4*)(W + (size_t)(n0 + row) * ldw + k0 + c * 4);
            }
        }
        if (t + 2 < nt) {  // gfx1250 global_prefetch_b8, 2 stages ahead on the weight stream
            const float* pf = W + (size_t)(n0 + (tid & 127)) * ldw + (t + 2) * BK + ((tid >> 7) & 1) * 32;
            __builtin_prefetch(pf, 0, 1);
        }
        const _Float16* cA = sA + cur * (BM * LDSP);
        const _Float16* cB = sB + cur * (BN * LDSP);
        mma_step(cA, cB, lane, mrow, ncol, 0, acc);
        mma_step(cA, cB, lane, mrow, ncol, 1, acc);
        if (t + 1 < nt) {
            stash_a(sA + (cur ^ 1) * (BM * LDSP), ra, tid);
            stash_b(sB + (cur ^ 1) * (BN * LDSP), rb, tid);
        }
        __syncthreads();
        cur ^= 1;
    }
}

// Kernel 1: t1[m, r] = (x @ V)[m, r] * S[r]   (M=512, N=RANK, K=K_IN)
__global__ void __launch_bounds__(THREADS, 2)
lora_xv_scaled(const float* __restrict__ x, const float* __restrict__ V,
               const float* __restrict__ S, float* __restrict__ t1) {
    __shared__ _Float16 sA[BM * LDSP];   // 18.4 KB
    __shared__ _Float16 sB[BN * LDSP];   // 18.4 KB
    const int tid = threadIdx.x, lane = tid & 31, wave = tid >> 5;
    const int mrow = (wave >> 2) * 64, ncol = (wave & 3) * 32;   // 2x4 wave grid, 64x32 tiles
    const int m0 = blockIdx.y * BM, n0 = blockIdx.x * BN;
    v8f acc[4][2] = {};

    for (int k0 = 0; k0 < K_IN; k0 += BK) {
        #pragma unroll
        for (int i = 0; i < NVA; ++i) {  // x tile, row-major
            int idx = tid + i * THREADS, row = idx >> 4, c = idx & 15;
            f4 v = *(const f4*)(x + (size_t)(m0 + row) * K_IN + k0 + c * 4);
            *(h4*)(sA + row * LDSP + c * 4) = __builtin_convertvector(v, h4);
        }
        #pragma unroll
        for (int i = 0; i < NVB; ++i) {  // V is [K, RANK]: coalesced read, transpose into sB[n][k]
            int idx = tid + i * THREADS, kr = idx >> 5, nc = idx & 31;
            f4 v = *(const f4*)(V + (size_t)(k0 + kr) * RANK + n0 + nc * 4);
            #pragma unroll
            for (int j = 0; j < 4; ++j)
                sB[(nc * 4 + j) * LDSP + kr] = (_Float16)v[j];
        }
        __syncthreads();
        mma_step(sA, sB, lane, mrow, ncol, 0, acc);
        mma_step(sA, sB, lane, mrow, ncol, 1, acc);
        __syncthreads();
    }
    #pragma unroll
    for (int fm = 0; fm < 4; ++fm)
        #pragma unroll
        for (int fn = 0; fn < 2; ++fn) {
            const int n = n0 + ncol + fn * 16 + (lane & 15);
            const float sv = S[n];
            #pragma unroll
            for (int i = 0; i < 8; ++i) {
                const int m = m0 + mrow + fm * 16 + ((lane >> 4) << 3) + i;
                t1[(size_t)m * RANK + n] = acc[fm][fn][i] * sv;
            }
        }
}

// Kernel 2: y = x @ base^T + t1 @ U^T  (concatenated-K fused GEMM, dynamic 72KB LDS)
__global__ void __launch_bounds__(THREADS)
delta_fused_gemm(const float* __restrict__ x, const float* __restrict__ base,
                 const float* __restrict__ U, const float* __restrict__ t1,
                 float* __restrict__ y) {
    extern __shared__ char smem_raw[];
    _Float16* sA = (_Float16*)smem_raw;                 // 2 * 128 * 72 halves
    _Float16* sB = sA + 2 * BM * LDSP;                  // 2 * 128 * 72 halves
    const int tid = threadIdx.x, lane = tid & 31, wave = tid >> 5;
    const int mrow = (wave >> 2) * 64, ncol = (wave & 3) * 32;   // 2x4 wave grid, 64x32 tiles
    // blockIdx.x = M tile (fast-varying): the 4 M-blocks per N-tile are adjacent in
    // launch order -> co-resident -> base tiles are fetched from HBM ~once, shared via L2.
    const int m0 = blockIdx.x * BM, n0 = blockIdx.y * BN;
    v8f acc[4][2] = {};

    gemm_pass(x,  K_IN, base, K_IN, K_IN, m0, n0, sA, sB, tid, lane, mrow, ncol, acc);
    gemm_pass(t1, RANK, U,    RANK, RANK, m0, n0, sA, sB, tid, lane, mrow, ncol, acc);

    #pragma unroll
    for (int fm = 0; fm < 4; ++fm)
        #pragma unroll
        for (int fn = 0; fn < 2; ++fn) {
            const int n = n0 + ncol + fn * 16 + (lane & 15);
            #pragma unroll
            for (int i = 0; i < 8; ++i) {
                const int m = m0 + mrow + fm * 16 + ((lane >> 4) << 3) + i;
                y[(size_t)m * N_OUT + n] = acc[fm][fn][i];
            }
        }
}

extern "C" void kernel_launch(void* const* d_in, const int* in_sizes, int n_in,
                              void* d_out, int out_size, void* d_ws, size_t ws_size,
                              hipStream_t stream) {
    const float* x    = (const float*)d_in[0];  // [512, 8192]
    const float* base = (const float*)d_in[1];  // [8192, 8192]
    const float* U    = (const float*)d_in[2];  // [8192, 1024]
    const float* S    = (const float*)d_in[3];  // [1024]
    const float* V    = (const float*)d_in[4];  // [8192, 1024]
    float* y  = (float*)d_out;                  // [512, 8192]
    float* t1 = (float*)d_ws;                   // [512, 1024] fp32 scratch (2 MB)

    lora_xv_scaled<<<dim3(RANK / BN, M_TOT / BM), THREADS, 0, stream>>>(x, V, S, t1);
    delta_fused_gemm<<<dim3(M_TOT / BM, N_OUT / BN), THREADS, SMEM_BYTES, stream>>>(x, base, U, t1, y);
}
